// TropicalLinear_12335146074807
// MI455X (gfx1250) — compile-verified, matrix operand order checked by reading, not run on hardware
//
#include <hip/hip_runtime.h>

typedef __attribute__((ext_vector_type(16))) _Float16 v16h;
typedef __attribute__((ext_vector_type(8)))  _Float16 v8h;
typedef __attribute__((ext_vector_type(4)))  _Float16 v4h;
typedef __attribute__((ext_vector_type(8)))  float    v8f;
typedef __attribute__((ext_vector_type(4)))  float    v4f;

#define B_DIM   1024
#define IN_DIM  512
#define OUT_DIM 512

// ---------------------------------------------------------------------------
// Phase 1: per-row max + exp(v - rowmax) -> f16.  One wave32 per row.
// cols fixed at IN_DIM=512 -> 16 floats per lane (4 x float4, coalesced).
// ---------------------------------------------------------------------------
__global__ __launch_bounds__(256) void rowmax_exp_f16_kernel(
    const float* __restrict__ in, _Float16* __restrict__ outh,
    float* __restrict__ rowmax, int rows)
{
    const int wave = threadIdx.x >> 5;
    const int lane = threadIdx.x & 31;
    const int row  = blockIdx.x * 8 + wave;
    if (row >= rows) return;   // rows % 8 == 0 -> never divergent

    const float* p = in + (size_t)row * IN_DIM + lane * 4;

    v4f vals[4];
    float m = -__builtin_inff();
#pragma unroll
    for (int j = 0; j < 4; ++j) {
        vals[j] = *(const v4f*)(p + j * 128);
#pragma unroll
        for (int c = 0; c < 4; ++c) m = fmaxf(m, vals[j][c]);
    }
    // wave32 max-reduce
#pragma unroll
    for (int off = 16; off >= 1; off >>= 1)
        m = fmaxf(m, __shfl_xor(m, off, 32));

    if (lane == 0) rowmax[row] = m;

    _Float16* q = outh + (size_t)row * IN_DIM + lane * 4;
#pragma unroll
    for (int j = 0; j < 4; ++j) {
        v4h h;
#pragma unroll
        for (int c = 0; c < 4; ++c)
            h[c] = (_Float16)__expf(vals[j][c] - m);
        *(v4h*)(q + j * 128) = h;
    }
}

// ---------------------------------------------------------------------------
// Phase 2: C[b,o] = sum_i Ex[b,i]*Ew[o,i] via v_wmma_f32_16x16x32_f16 with
// 2x2 register blocking (wave -> 32x32 C region, 4 accumulators), then
// y = log(C) + mx[b] + mw[o] + bias[o].
// Per k-step: 2 A-frags + 2 B-frags (8 x b128 loads) feed 4 WMMAs
// (independent accumulators -> no D->A/B hazard).
// A-operand lane layout (16-bit 16x32): lane m    -> row m, K {0..7,16..23}
//                                       lane m+16 -> row m, K {8..15,24..31}
// B-operand: identical pattern over columns N -> load Ew rows the same way.
// ---------------------------------------------------------------------------
__global__ __launch_bounds__(256) void tropical_gemm_wmma_kernel(
    const _Float16* __restrict__ Ex, const _Float16* __restrict__ Ew,
    const float* __restrict__ mx, const float* __restrict__ mw,
    const float* __restrict__ bias, float* __restrict__ y)
{
    const int NS    = OUT_DIM / 32;            // 16 N super-tiles
    const int wave  = threadIdx.x >> 5;
    const int gw    = blockIdx.x * 8 + wave;   // global wave id
    const int bSup  = gw / NS;                 // 32-row M block
    const int oSup  = gw % NS;                 // 32-col N block
    const int lane  = threadIdx.x & 31;
    const int hsel  = lane >> 4;               // 0: K-low half, 1: K-high half
    const int r     = lane & 15;               // fragment row (A) / column (B)
    const int koff  = hsel * 8;

    const _Float16* aptr0 = Ex + (size_t)(bSup * 32 + r) * IN_DIM + koff;
    const _Float16* aptr1 = aptr0 + (size_t)16 * IN_DIM;
    const _Float16* bptr0 = Ew + (size_t)(oSup * 32 + r) * IN_DIM + koff;
    const _Float16* bptr1 = bptr0 + (size_t)16 * IN_DIM;

    v8f c00 = {}, c01 = {}, c10 = {}, c11 = {};
#pragma unroll
    for (int k = 0; k < IN_DIM; k += 32) {
        v8h a0l = *(const v8h*)(aptr0 + k);
        v8h a0h = *(const v8h*)(aptr0 + k + 16);
        v8h a1l = *(const v8h*)(aptr1 + k);
        v8h a1h = *(const v8h*)(aptr1 + k + 16);
        v8h b0l = *(const v8h*)(bptr0 + k);
        v8h b0h = *(const v8h*)(bptr0 + k + 16);
        v8h b1l = *(const v8h*)(bptr1 + k);
        v8h b1h = *(const v8h*)(bptr1 + k + 16);
        v16h A0 = __builtin_shufflevector(a0l, a0h, 0,1,2,3,4,5,6,7,8,9,10,11,12,13,14,15);
        v16h A1 = __builtin_shufflevector(a1l, a1h, 0,1,2,3,4,5,6,7,8,9,10,11,12,13,14,15);
        v16h B0 = __builtin_shufflevector(b0l, b0h, 0,1,2,3,4,5,6,7,8,9,10,11,12,13,14,15);
        v16h B1 = __builtin_shufflevector(b1l, b1h, 0,1,2,3,4,5,6,7,8,9,10,11,12,13,14,15);
        c00 = __builtin_amdgcn_wmma_f32_16x16x32_f16(false, A0, false, B0, (short)0, c00, false, false);
        c01 = __builtin_amdgcn_wmma_f32_16x16x32_f16(false, A0, false, B1, (short)0, c01, false, false);
        c10 = __builtin_amdgcn_wmma_f32_16x16x32_f16(false, A1, false, B0, (short)0, c10, false, false);
        c11 = __builtin_amdgcn_wmma_f32_16x16x32_f16(false, A1, false, B1, (short)0, c11, false, false);
    }

    // C layout per tile: VGPR e, lanes 0-15 -> M=e, N=lane;
    //                    lanes 16-31 -> M=e+8, N=lane-16
    const int n     = lane & 15;
    const int mbase = hsel * 8;
    const int ocol0 = oSup * 32 + n;          // N tile 0
    const int ocol1 = ocol0 + 16;             // N tile 1
    const float addo0 = mw[ocol0] + bias[ocol0];
    const float addo1 = mw[ocol1] + bias[ocol1];

    v8f cc[4] = { c00, c01, c10, c11 };
#pragma unroll
    for (int tm = 0; tm < 2; ++tm) {
#pragma unroll
        for (int e = 0; e < 8; ++e) {
            const int brow = bSup * 32 + tm * 16 + mbase + e;
            const float mxv = mx[brow];
            float* yrow = y + (size_t)brow * OUT_DIM;
            yrow[ocol0] = __logf(cc[tm * 2 + 0][e]) + mxv + addo0;
            yrow[ocol1] = __logf(cc[tm * 2 + 1][e]) + mxv + addo1;
        }
    }
}

// ---------------------------------------------------------------------------
extern "C" void kernel_launch(void* const* d_in, const int* in_sizes, int n_in,
                              void* d_out, int out_size, void* d_ws, size_t ws_size,
                              hipStream_t stream)
{
    const float* x    = (const float*)d_in[0];   // [1024, 512]
    const float* w    = (const float*)d_in[1];   // [512, 512]
    const float* bias = (const float*)d_in[2];   // [512]
    float*       y    = (float*)d_out;           // [1024, 512]

    char* ws = (char*)d_ws;
    _Float16* Ex = (_Float16*)ws;                                   // 1 MiB
    _Float16* Ew = (_Float16*)(ws + (size_t)B_DIM * IN_DIM * 2);    // 512 KiB
    float*    mx = (float*)(ws + (size_t)(B_DIM + OUT_DIM) * IN_DIM * 2);
    float*    mw = mx + B_DIM;

    rowmax_exp_f16_kernel<<<B_DIM / 8, 256, 0, stream>>>(x, Ex, mx, B_DIM);
    rowmax_exp_f16_kernel<<<OUT_DIM / 8, 256, 0, stream>>>(w, Ew, mw, OUT_DIM);

    const int totalWaves = (B_DIM / 32) * (OUT_DIM / 32);  // 512 waves
    tropical_gemm_wmma_kernel<<<totalWaves / 8, 256, 0, stream>>>(Ex, Ew, mx, mw, bias, y);
}
